// Qwen3_5MoE_39874476376659
// MI455X (gfx1250) — matrix operand order repaired and checked by reading, not compiled, not run on hardware
//
#include <hip/hip_runtime.h>
#include <hip/hip_bf16.h>

#define T_TOK 2048
#define HID   1024
#define MID   512
#define NEXP  32
#define TOPK  4
#define CAP   2048   // per-expert worst-case token capacity

typedef __attribute__((ext_vector_type(16))) __bf16 v16bf;
typedef __attribute__((ext_vector_type(8)))  float  v8f;

union Frag {
    v16bf        v;
    uint4        q[2];
    unsigned int u[8];
};

static __device__ __forceinline__ unsigned short f2bf(float f) {
    union { __bf16 b; unsigned short u; } c;
    c.b = (__bf16)f;
    return c.u;
}

static __device__ __forceinline__ unsigned int pack2(float a, float b) {
    return (unsigned int)f2bf(a) | ((unsigned int)f2bf(b) << 16);
}

// Load 16 consecutive fp32, convert to a packed bf16 WMMA fragment half.
static __device__ __forceinline__ Frag load16f(const float* __restrict__ p) {
    float4 f0 = ((const float4*)p)[0];
    float4 f1 = ((const float4*)p)[1];
    float4 f2 = ((const float4*)p)[2];
    float4 f3 = ((const float4*)p)[3];
    Frag r;
    r.u[0] = pack2(f0.x, f0.y); r.u[1] = pack2(f0.z, f0.w);
    r.u[2] = pack2(f1.x, f1.y); r.u[3] = pack2(f1.z, f1.w);
    r.u[4] = pack2(f2.x, f2.y); r.u[5] = pack2(f2.z, f2.w);
    r.u[6] = pack2(f3.x, f3.y); r.u[7] = pack2(f3.z, f3.w);
    return r;
}

// gfx1250 async DMA: global -> LDS, 16 bytes per lane (tracked by ASYNCcnt).
// LDS offset = low 32 bits of the generic pointer (flat LDS aperture rule).
static __device__ __forceinline__ void async_ld_b128(void* lds_dst, const void* gsrc) {
    unsigned            loff = (unsigned)(unsigned long long)lds_dst;
    unsigned long long  ga   = (unsigned long long)gsrc;
    asm volatile("global_load_async_to_lds_b128 %0, %1, off"
                 :: "v"(loff), "v"(ga) : "memory");
}
static __device__ __forceinline__ void wait_async0() {
    asm volatile("s_wait_asynccnt 0" ::: "memory");
}

static __device__ __forceinline__ float fast_silu(float g) {
    return g * __builtin_amdgcn_rcpf(1.f + __expf(-g));
}

// ---------------------------------------------------------------------------
// Kernel 0: convert activations fp32 -> bf16 once (A-matrices become bf16).
// ---------------------------------------------------------------------------
__global__ __launch_bounds__(256) void cvt_x_kernel(
    const float* __restrict__ x, unsigned short* __restrict__ xbf)
{
    int i = blockIdx.x * 256 + threadIdx.x;      // one 8-element group each
    const float* p = x + (size_t)i * 8;
    float4 f0 = ((const float4*)p)[0];
    float4 f1 = ((const float4*)p)[1];
    uint4 o;
    o.x = pack2(f0.x, f0.y); o.y = pack2(f0.z, f0.w);
    o.z = pack2(f1.x, f1.y); o.w = pack2(f1.z, f1.w);
    *(uint4*)(xbf + (size_t)i * 8) = o;
}

// ---------------------------------------------------------------------------
// Kernel 1: routing. One wave32 per token; lane == expert id (E == 32).
// ---------------------------------------------------------------------------
__global__ __launch_bounds__(32) void route_kernel(
    const float* __restrict__ x, const float* __restrict__ gw,
    int* __restrict__ counts, int* __restrict__ etok,
    int* __restrict__ eslot, float* __restrict__ escale)
{
    int t    = blockIdx.x;
    int lane = threadIdx.x;             // expert id
    const float* xr = x  + (size_t)t    * HID;
    const float* gr = gw + (size_t)lane * HID;

    float acc = 0.f;
    #pragma unroll 4
    for (int i = 0; i < HID; i += 4) {
        float4 a = *(const float4*)(xr + i);
        float4 b = *(const float4*)(gr + i);
        acc = fmaf(a.x, b.x, acc);
        acc = fmaf(a.y, b.y, acc);
        acc = fmaf(a.z, b.z, acc);
        acc = fmaf(a.w, b.w, acc);
    }

    // top-4 across the wave, deterministic tie-break on smaller index
    float v = acc;
    int   idx = lane;
    float selv[TOPK];
    int   seli[TOPK];
    #pragma unroll
    for (int k = 0; k < TOPK; ++k) {
        float mv = v; int mi = idx;
        #pragma unroll
        for (int off = 16; off >= 1; off >>= 1) {
            float ov = __shfl_xor(mv, off, 32);
            int   oi = __shfl_xor(mi, off, 32);
            if (ov > mv || (ov == mv && oi < mi)) { mv = ov; mi = oi; }
        }
        selv[k] = mv; seli[k] = mi;
        if (lane == mi) v = -3.0e38f;
    }

    float s = 0.f, w[TOPK];
    #pragma unroll
    for (int k = 0; k < TOPK; ++k) { w[k] = __expf(selv[k] - selv[0]); s += w[k]; }

    if (lane < TOPK) {
        int e   = seli[lane];
        int pos = atomicAdd(&counts[e], 1);
        etok[e * CAP + pos]   = t;
        eslot[e * CAP + pos]  = lane;
        escale[e * CAP + pos] = w[lane] / s;
    }
}

// ---------------------------------------------------------------------------
// Kernel 2: gate & up projections + silu + route scaling -> mixed (bf16).
// grid: (expert, 64-token tile, 128-m block), 256 threads (8 waves).
// Each wave: four 16x16 row-tiles x (gate,up) = 8 accumulators; B reused 4x.
// ---------------------------------------------------------------------------
__global__ __launch_bounds__(256) void gate_up_kernel(
    const unsigned short* __restrict__ xbf, const float* __restrict__ w_gate,
    const float* __restrict__ w_up, const int* __restrict__ counts,
    const int* __restrict__ etok, const float* __restrict__ escale,
    unsigned short* __restrict__ mixed)
{
    int e   = blockIdx.x;
    int cnt = counts[e];
    int tileBase = blockIdx.y * 64;
    if (tileBase >= cnt) return;

    // X tile: 64 tokens x 1024 bf16, padded row stride 1032 (bank-conflict free)
    __shared__ unsigned short lx[64 * 1032];
    int tid = threadIdx.x;
    for (int i = tid; i < 64 * 128; i += 256) {   // 128 x 16B chunks per row
        int r = i >> 7;
        int c = (i & 127) << 3;                   // bf16 element offset
        int g = tileBase + r;
        if (g < cnt) {
            async_ld_b128(&lx[r * 1032 + c],
                          xbf + (size_t)etok[e * CAP + g] * HID + c);
        } else {
            *(uint4*)(&lx[r * 1032 + c]) = make_uint4(0u, 0u, 0u, 0u);
        }
    }
    wait_async0();
    __syncthreads();

    int wave = tid >> 5, lane = tid & 31;
    int col  = lane & 15;
    bool hi  = lane >= 16;
    int m0   = blockIdx.z * 128 + wave * 16;

    const float* wg = w_gate + ((size_t)e * MID + m0 + col) * HID;
    const float* wu = w_up   + ((size_t)e * MID + m0 + col) * HID;
    const unsigned short* la[4];
    #pragma unroll
    for (int rt = 0; rt < 4; ++rt) la[rt] = &lx[(rt * 16 + col) * 1032];

    v8f cg[4] = {{}, {}, {}, {}};
    v8f cu[4] = {{}, {}, {}, {}};
    int aoff = hi ? 16 : 0;   // A: 16 contiguous K starting at 16*(lane>=16)
    int boff = hi ? 16 : 0;   // B: 16 contiguous K starting at 16*(lane>=16)

    for (int k0 = 0; k0 < HID; k0 += 32) {
        Frag a[4];
        #pragma unroll
        for (int rt = 0; rt < 4; ++rt) {
            a[rt].q[0] = *(const uint4*)(la[rt] + k0 + aoff);
            a[rt].q[1] = *(const uint4*)(la[rt] + k0 + aoff + 8);
        }
        Frag bg = load16f(wg + k0 + boff);
        Frag bu = load16f(wu + k0 + boff);
        #pragma unroll
        for (int rt = 0; rt < 4; ++rt) {
            cg[rt] = __builtin_amdgcn_wmma_f32_16x16x32_bf16(
                false, a[rt].v, false, bg.v, (short)0, cg[rt], false, false);
            cu[rt] = __builtin_amdgcn_wmma_f32_16x16x32_bf16(
                false, a[rt].v, false, bu.v, (short)0, cu[rt], false, false);
        }
    }

    int rbase = (hi ? 8 : 0);
    #pragma unroll
    for (int rt = 0; rt < 4; ++rt) {
        #pragma unroll
        for (int j = 0; j < 8; ++j) {
            int g = tileBase + rt * 16 + rbase + j;
            if (g < cnt) {
                float sc  = escale[e * CAP + g];
                float val = fast_silu(cg[rt][j]) * cu[rt][j] * sc;
                mixed[((size_t)e * CAP + g) * MID + m0 + col] = f2bf(val);
            }
        }
    }
}

// ---------------------------------------------------------------------------
// Kernel 3: down projection -> per-(token,slot) f32 partial contributions.
// grid: (expert, 64-token tile, 128-h block), 256 threads (8 waves).
// ---------------------------------------------------------------------------
__global__ __launch_bounds__(256) void down_kernel(
    const unsigned short* __restrict__ mixed, const float* __restrict__ w_down,
    const int* __restrict__ counts, const int* __restrict__ etok,
    const int* __restrict__ eslot, float* __restrict__ partial)
{
    int e   = blockIdx.x;
    int cnt = counts[e];
    int tileBase = blockIdx.y * 64;
    if (tileBase >= cnt) return;

    // mixed tile: 64 tokens x 512 bf16, padded stride 520
    __shared__ unsigned short lm[64 * 520];
    int tid = threadIdx.x;
    for (int i = tid; i < 64 * 64; i += 256) {    // 64 x 16B chunks per row
        int r = i >> 6;
        int c = (i & 63) << 3;
        int g = tileBase + r;
        if (g < cnt) {
            async_ld_b128(&lm[r * 520 + c],
                          mixed + ((size_t)e * CAP + g) * MID + c);
        } else {
            *(uint4*)(&lm[r * 520 + c]) = make_uint4(0u, 0u, 0u, 0u);
        }
    }
    wait_async0();
    __syncthreads();

    int wave = tid >> 5, lane = tid & 31;
    int col  = lane & 15;
    bool hi  = lane >= 16;
    int h0   = blockIdx.z * 128 + wave * 16;

    const float* wd = w_down + ((size_t)e * HID + h0 + col) * MID;
    const unsigned short* la[4];
    #pragma unroll
    for (int rt = 0; rt < 4; ++rt) la[rt] = &lm[(rt * 16 + col) * 520];

    v8f cacc[4] = {{}, {}, {}, {}};
    int aoff = hi ? 16 : 0;
    int boff = hi ? 16 : 0;

    for (int k0 = 0; k0 < MID; k0 += 32) {
        Frag a[4];
        #pragma unroll
        for (int rt = 0; rt < 4; ++rt) {
            a[rt].q[0] = *(const uint4*)(la[rt] + k0 + aoff);
            a[rt].q[1] = *(const uint4*)(la[rt] + k0 + aoff + 8);
        }
        Frag b = load16f(wd + k0 + boff);
        #pragma unroll
        for (int rt = 0; rt < 4; ++rt) {
            cacc[rt] = __builtin_amdgcn_wmma_f32_16x16x32_bf16(
                false, a[rt].v, false, b.v, (short)0, cacc[rt], false, false);
        }
    }

    int rbase = (hi ? 8 : 0);
    #pragma unroll
    for (int rt = 0; rt < 4; ++rt) {
        #pragma unroll
        for (int j = 0; j < 8; ++j) {
            int g = tileBase + rt * 16 + rbase + j;
            if (g < cnt) {
                int t  = etok[e * CAP + g];
                int sl = eslot[e * CAP + g];
                partial[(((size_t)t * TOPK) + sl) * HID + h0 + col] = cacc[rt][j];
            }
        }
    }
}

// ---------------------------------------------------------------------------
// Kernel 4: fixed-order reduction of the 4 slot contributions per token.
// ---------------------------------------------------------------------------
__global__ __launch_bounds__(256) void reduce_kernel(
    const float* __restrict__ partial, float* __restrict__ out)
{
    int i = blockIdx.x * 256 + threadIdx.x;          // i in [0, T*H)
    const float* p = partial + ((size_t)(i >> 10) * TOPK) * HID + (i & 1023);
    out[i] = (p[0] + p[HID]) + (p[2 * HID] + p[3 * HID]);
}

// ---------------------------------------------------------------------------
extern "C" void kernel_launch(void* const* d_in, const int* in_sizes, int n_in,
                              void* d_out, int out_size, void* d_ws, size_t ws_size,
                              hipStream_t stream)
{
    (void)in_sizes; (void)n_in; (void)out_size; (void)ws_size;
    const float* x      = (const float*)d_in[0];   // (T,1,H)
    const float* gate_w = (const float*)d_in[1];   // (E,H)
    const float* w_gate = (const float*)d_in[2];   // (E,M,H)
    const float* w_up   = (const float*)d_in[3];   // (E,M,H)
    const float* w_down = (const float*)d_in[4];   // (E,H,M)
    float* out = (float*)d_out;                    // (T,1,H)

    char* ws = (char*)d_ws;
    size_t off = 0;
    auto alloc = [&](size_t bytes) -> void* {
        void* p = ws + off;
        off = (off + bytes + 255) & ~(size_t)255;
        return p;
    };
    int*            counts  = (int*)alloc((size_t)NEXP * sizeof(int));
    int*            etok    = (int*)alloc((size_t)NEXP * CAP * sizeof(int));
    int*            eslot   = (int*)alloc((size_t)NEXP * CAP * sizeof(int));
    float*          escale  = (float*)alloc((size_t)NEXP * CAP * sizeof(float));
    unsigned short* xbf     = (unsigned short*)alloc((size_t)T_TOK * HID * 2);
    unsigned short* mixed   = (unsigned short*)alloc((size_t)NEXP * CAP * MID * 2);
    float*          partial = (float*)alloc((size_t)T_TOK * TOPK * HID * sizeof(float));

    hipMemsetAsync(counts, 0, NEXP * sizeof(int), stream);

    cvt_x_kernel<<<(T_TOK * HID / 8) / 256, 256, 0, stream>>>(x, xbf);

    route_kernel<<<T_TOK, 32, 0, stream>>>(x, gate_w, counts, etok, eslot, escale);

    gate_up_kernel<<<dim3(NEXP, CAP / 64, MID / 128), 256, 0, stream>>>(
        xbf, w_gate, w_up, counts, etok, escale, mixed);

    down_kernel<<<dim3(NEXP, CAP / 64, HID / 128), 256, 0, stream>>>(
        mixed, w_down, counts, etok, eslot, partial);

    reduce_kernel<<<(T_TOK * HID) / 256, 256, 0, stream>>>(partial, out);
}
